// ChiralMessage_70317204570248
// MI455X (gfx1250) — matrix operand order repaired and checked by reading, not compiled, not run on hardware
//
#include <hip/hip_runtime.h>

typedef __attribute__((ext_vector_type(16))) _Float16 v16h;
typedef __attribute__((ext_vector_type(8)))  _Float16 v8h;
typedef __attribute__((ext_vector_type(8)))  float    v8f;

#define WAVES 8
#define TPW   4

__device__ __forceinline__ v8f wmma16(v16h a, v16h b, v8f c) {
  // D = A(16x32 f16) * B(32x16 f16) + C(16x16 f32)
  return __builtin_amdgcn_wmma_f32_16x16x32_f16(false, a, false, b, (short)0, c, false, false);
}

__device__ __forceinline__ float silu_f(float x) { return x / (1.0f + __expf(-x)); }

// A-fragment (16x32 f16) from a 32-float global row chunk.
// ISA layout: lane<16 -> K {0..7, 16..23}; lane>=16 -> K {8..15, 24..31}
__device__ __forceinline__ v16h load_a_global(const float* rowChunk, int laneHi) {
  const float4* p = (const float4*)rowChunk;
  int o = laneHi ? 2 : 0;
  float4 x0 = p[o + 0], x1 = p[o + 1], x2 = p[o + 4], x3 = p[o + 5];
  v16h a;
  a[0]=(_Float16)x0.x;  a[1]=(_Float16)x0.y;  a[2]=(_Float16)x0.z;  a[3]=(_Float16)x0.w;
  a[4]=(_Float16)x1.x;  a[5]=(_Float16)x1.y;  a[6]=(_Float16)x1.z;  a[7]=(_Float16)x1.w;
  a[8]=(_Float16)x2.x;  a[9]=(_Float16)x2.y;  a[10]=(_Float16)x2.z; a[11]=(_Float16)x2.w;
  a[12]=(_Float16)x3.x; a[13]=(_Float16)x3.y; a[14]=(_Float16)x3.z; a[15]=(_Float16)x3.w;
  return a;
}

// A-fragment from a 16x64 f16 LDS tile (row-major), K-chunk kc.
__device__ __forceinline__ v16h load_a_lds(const _Float16* h0, int lr, int laneHi, int kc) {
  const _Float16* p = h0 + lr * 64 + kc * 32 + (laneHi ? 8 : 0);
  v8h lo = *(const v8h*)p;
  v8h hi = *(const v8h*)(p + 16);
  v16h a;
#pragma unroll
  for (int j = 0; j < 8; ++j) { a[j] = lo[j]; a[8 + j] = hi[j]; }
  return a;
}

// Pre-swizzled B fragment: 32 lanes x 16 f16, contiguous per fragment.
__device__ __forceinline__ v16h load_b_lds(const _Float16* sB, int frag, int ln) {
  return *(const v16h*)(sB + (frag * 32 + ln) * 16);
}

// Stage W (row-major fanIn x nCols, f32) into LDS in B-fragment order.
// Fragment f = kc*numNt + nt; element (ln, j) = W[32*kc + 8*(ln>>4) + (j&7) + 16*(j>>3)][16*nt + (ln&15)]
__device__ __forceinline__ void stage_weights(const float* W, _Float16* sB,
                                              int nCols, int numNt, int totalFrags) {
  int total = totalFrags * 512;
  for (int i = threadIdx.x; i < total; i += blockDim.x) {
    int f = i >> 9, r = i & 511, ln = r >> 4, j = r & 15;
    int kc = f / numNt, nt = f - kc * numNt;
    int k = 32 * kc + 8 * (ln >> 4) + (j & 7) + 16 * (j >> 3);
    int n = 16 * nt + (ln & 15);
    sB[i] = (_Float16)W[k * nCols + n];
  }
}

// ---------------- Edge kernel: MLP + gated messages + segment sums ----------------
__global__ void __launch_bounds__(256)
k_edge(const float* __restrict__ ns, const float* __restrict__ nv,
       const float* __restrict__ pos, const long long* __restrict__ eidx,
       const float* __restrict__ Ws1, const float* __restrict__ bs1,
       const float* __restrict__ Ws2, const float* __restrict__ bs2,
       float* __restrict__ out1, float* __restrict__ out3, int numTiles) {
  __shared__ __align__(32) _Float16 sB1[16 * 512];   // Ws1: 4 kc x 4 nt
  __shared__ __align__(32) _Float16 sB2[24 * 512];   // Ws2: 2 kc x 12 nt
  __shared__ __align__(32) _Float16 sH[WAVES][1024]; // per-wave 16x64 staging
  __shared__ float sb1[64];
  __shared__ float sb2[192];
  __shared__ float sRel[WAVES][48];
  __shared__ int   sE0[WAVES][16];
  __shared__ int   sE1[WAVES][16];

  stage_weights(Ws1, sB1, 64, 4, 16);
  stage_weights(Ws2, sB2, 192, 12, 24);
  if (threadIdx.x < 64)  sb1[threadIdx.x] = bs1[threadIdx.x];
  if (threadIdx.x < 192) sb2[threadIdx.x] = bs2[threadIdx.x];
  __syncthreads();

  int wave = threadIdx.x >> 5, ln = threadIdx.x & 31;
  int laneHi = ln >> 4, lr = ln & 15;

  for (int it = 0; it < TPW; ++it) {
    int tile = blockIdx.x * (WAVES * TPW) + it * WAVES + wave;
    bool act = tile < numTiles;                 // wave-uniform
    if (act) {
      int edge = tile * 16 + lr;
      int e0 = (int)eidx[2 * edge + 0];
      int e1 = (int)eidx[2 * edge + 1];
      if (ln < 16) {
        sE0[wave][lr] = e0;
        sE1[wave][lr] = e1;
#pragma unroll
        for (int ax = 0; ax < 3; ++ax)
          sRel[wave][lr * 3 + ax] = pos[e1 * 3 + ax] - pos[e0 * 3 + ax];
      }
      // ---- layer 1: [s[e0]|s[e1]](16x128) @ Ws1(128x64) ----
      v8f h[4];
#pragma unroll
      for (int nt = 0; nt < 4; ++nt) h[nt] = (v8f){};
#pragma unroll
      for (int kc = 0; kc < 4; ++kc) {
        const float* rc = ns + (size_t)(kc < 2 ? e0 : e1) * 64 + (kc & 1) * 32;
        v16h a = load_a_global(rc, laneHi);
#pragma unroll
        for (int nt = 0; nt < 4; ++nt)
          h[nt] = wmma16(a, load_b_lds(sB1, kc * 4 + nt, ln), h[nt]);
      }
      // silu -> f16 staging tile
#pragma unroll
      for (int nt = 0; nt < 4; ++nt) {
        float b = sb1[16 * nt + lr];
#pragma unroll
        for (int r = 0; r < 8; ++r) {
          float x = h[nt][r] + b;
          sH[wave][(r + 8 * laneHi) * 64 + 16 * nt + lr] = (_Float16)silu_f(x);
        }
      }
    }
    __syncthreads();
    if (act) {
      // ---- layer 2: h(16x64) @ Ws2(64x192) ----
      v8f g[12];
#pragma unroll
      for (int nt = 0; nt < 12; ++nt) g[nt] = (v8f){};
#pragma unroll
      for (int kc = 0; kc < 2; ++kc) {
        v16h a = load_a_lds(&sH[wave][0], lr, laneHi, kc);
#pragma unroll
        for (int nt = 0; nt < 12; ++nt)
          g[nt] = wmma16(a, load_b_lds(sB2, kc * 12 + nt, ln), g[nt]);
      }
      // gate_vector = g[:, :64], gate_edge = g[:, 64:128], messages_ss = g[:, 128:192]
#pragma unroll
      for (int nt = 0; nt < 4; ++nt) {
        int col = 16 * nt + lr;
        float bgv = sb2[col], bge = sb2[64 + col], bss = sb2[128 + col];
#pragma unroll
        for (int r = 0; r < 8; ++r) {
          int row = r + 8 * laneHi;
          int te0 = sE0[wave][row], te1 = sE1[wave][row];
          float gv = g[nt][r] + bgv;
          float ge = g[4 + nt][r] + bge;
          float ms = g[8 + nt][r] + bss;
          atomicAdd(&out1[(size_t)te0 * 64 + col], ms);
#pragma unroll
          for (int ax = 0; ax < 3; ++ax) {
            float nvv = nv[(size_t)te1 * 192 + ax * 64 + col];
            atomicAdd(&out3[(size_t)te0 * 192 + ax * 64 + col],
                      gv * nvv + ge * sRel[wave][row * 3 + ax]);
          }
        }
      }
    }
    __syncthreads();
  }
}

// ---------------- Vector-norm gate: out1 += node_scalar * ||node_vector@WV + bV|| ----------------
__global__ void __launch_bounds__(256)
k_vs(const float* __restrict__ ns, const float* __restrict__ nv,
     const float* __restrict__ WV, const float* __restrict__ bV,
     float* __restrict__ out1, int numTiles) {
  __shared__ __align__(32) _Float16 sB[8 * 512]; // WV: 2 kc x 4 nt
  __shared__ float sbv[64];
  stage_weights(WV, sB, 64, 4, 8);
  if (threadIdx.x < 64) sbv[threadIdx.x] = bV[threadIdx.x];
  __syncthreads();

  int wave = threadIdx.x >> 5, ln = threadIdx.x & 31;
  int laneHi = ln >> 4, lr = ln & 15;
  int tile = blockIdx.x * WAVES + wave;
  if (tile >= numTiles) return;
  int n0 = tile * 16;

  v16h a[3][2];
#pragma unroll
  for (int ax = 0; ax < 3; ++ax)
#pragma unroll
    for (int kc = 0; kc < 2; ++kc)
      a[ax][kc] = load_a_global(nv + (size_t)(n0 + lr) * 192 + ax * 64 + kc * 32, laneHi);

  v8f sq[4];
#pragma unroll
  for (int nt = 0; nt < 4; ++nt) sq[nt] = (v8f){};
#pragma unroll
  for (int nt = 0; nt < 4; ++nt) {
    float b = sbv[16 * nt + lr];
#pragma unroll
    for (int ax = 0; ax < 3; ++ax) {
      v8f c = (v8f){};
#pragma unroll
      for (int kc = 0; kc < 2; ++kc)
        c = wmma16(a[ax][kc], load_b_lds(sB, kc * 4 + nt, ln), c);
#pragma unroll
      for (int r = 0; r < 8; ++r) { float v = c[r] + b; sq[nt][r] += v * v; }
    }
  }
#pragma unroll
  for (int nt = 0; nt < 4; ++nt) {
#pragma unroll
    for (int r = 0; r < 8; ++r) {
      int node = n0 + r + 8 * laneHi;
      int col = 16 * nt + lr;
      size_t off = (size_t)node * 64 + col;
      out1[off] += ns[off] * sqrtf(sq[nt][r]);   // safe: separate launch, unique (node,col)
    }
  }
}

// ---------------- Triplet kernel: chiral gates + scalar-triple-product messages ----------------
__global__ void __launch_bounds__(256)
k_tri(const float* __restrict__ nc, const float* __restrict__ pos,
      const long long* __restrict__ tix,
      const float* __restrict__ Wc1, const float* __restrict__ bc1,
      const float* __restrict__ Wc2, const float* __restrict__ bc2,
      float* __restrict__ out2, int numTiles) {
  __shared__ __align__(32) _Float16 sB1[16 * 512];   // Wc1: 4 kc x 4 nt
  __shared__ __align__(32) _Float16 sB2[8 * 512];    // Wc2: 2 kc x 4 nt
  __shared__ __align__(32) _Float16 sH[WAVES][1024];
  __shared__ float sbc1[64], sbc2[64];
  __shared__ float sInv[WAVES][16];
  __shared__ int   sBase[WAVES][16];

  stage_weights(Wc1, sB1, 64, 4, 16);
  stage_weights(Wc2, sB2, 64, 4, 8);
  if (threadIdx.x < 64) { sbc1[threadIdx.x] = bc1[threadIdx.x]; sbc2[threadIdx.x] = bc2[threadIdx.x]; }
  __syncthreads();

  int wave = threadIdx.x >> 5, ln = threadIdx.x & 31;
  int laneHi = ln >> 4, lr = ln & 15;

  for (int it = 0; it < TPW; ++it) {
    int tile = blockIdx.x * (WAVES * TPW) + it * WAVES + wave;
    bool act = tile < numTiles;
    if (act) {
      int trip = tile * 16 + lr;
      int tb = (int)tix[4 * trip + 0];
      int t1 = (int)tix[4 * trip + 1];
      int t2 = (int)tix[4 * trip + 2];
      int t3 = (int)tix[4 * trip + 3];
      if (ln < 16) {
        float pbx = pos[tb * 3 + 0], pby = pos[tb * 3 + 1], pbz = pos[tb * 3 + 2];
        float r1x = pbx - pos[t1 * 3 + 0], r1y = pby - pos[t1 * 3 + 1], r1z = pbz - pos[t1 * 3 + 2];
        float r2x = pbx - pos[t2 * 3 + 0], r2y = pby - pos[t2 * 3 + 1], r2z = pbz - pos[t2 * 3 + 2];
        float r3x = pbx - pos[t3 * 3 + 0], r3y = pby - pos[t3 * 3 + 1], r3z = pbz - pos[t3 * 3 + 2];
        float cx = r2y * r3z - r2z * r3y;
        float cy = r2z * r3x - r2x * r3z;
        float cz = r2x * r3y - r2y * r3x;
        float stp = r1x * cx + r1y * cy + r1z * cz;
        sInv[wave][lr] = 1.0f / (stp + 0.01f);
        sBase[wave][lr] = tb;
      }
      // Layer-1 base half (cols 0..63 of concat input) — shared across the 3 branches
      v8f pb4[4];
#pragma unroll
      for (int nt = 0; nt < 4; ++nt) pb4[nt] = (v8f){};
#pragma unroll
      for (int kc = 0; kc < 2; ++kc) {
        v16h a = load_a_global(nc + (size_t)tb * 64 + kc * 32, laneHi);
#pragma unroll
        for (int nt = 0; nt < 4; ++nt)
          pb4[nt] = wmma16(a, load_b_lds(sB1, kc * 4 + nt, ln), pb4[nt]);
      }
      // hsum = sum_i silu(layer1_i); layer2 applied once to hsum
      v8f hs[4];
#pragma unroll
      for (int nt = 0; nt < 4; ++nt) hs[nt] = (v8f){};
      int others[3] = {t1, t2, t3};
#pragma unroll
      for (int br = 0; br < 3; ++br) {
        v8f c[4];
#pragma unroll
        for (int nt = 0; nt < 4; ++nt) c[nt] = pb4[nt];
#pragma unroll
        for (int kc = 2; kc < 4; ++kc) {
          v16h a = load_a_global(nc + (size_t)others[br] * 64 + (kc - 2) * 32, laneHi);
#pragma unroll
          for (int nt = 0; nt < 4; ++nt)
            c[nt] = wmma16(a, load_b_lds(sB1, kc * 4 + nt, ln), c[nt]);
        }
#pragma unroll
        for (int nt = 0; nt < 4; ++nt) {
          float b = sbc1[16 * nt + lr];
#pragma unroll
          for (int r = 0; r < 8; ++r) hs[nt][r] += silu_f(c[nt][r] + b);
        }
      }
#pragma unroll
      for (int nt = 0; nt < 4; ++nt)
#pragma unroll
        for (int r = 0; r < 8; ++r)
          sH[wave][(r + 8 * laneHi) * 64 + 16 * nt + lr] = (_Float16)hs[nt][r];
    }
    __syncthreads();
    if (act) {
      v8f g[4];
#pragma unroll
      for (int nt = 0; nt < 4; ++nt) g[nt] = (v8f){};
#pragma unroll
      for (int kc = 0; kc < 2; ++kc) {
        v16h a = load_a_lds(&sH[wave][0], lr, laneHi, kc);
#pragma unroll
        for (int nt = 0; nt < 4; ++nt)
          g[nt] = wmma16(a, load_b_lds(sB2, kc * 4 + nt, ln), g[nt]);
      }
#pragma unroll
      for (int nt = 0; nt < 4; ++nt) {
        int col = 16 * nt + lr;
        float b2 = 3.0f * sbc2[col];
#pragma unroll
        for (int r = 0; r < 8; ++r) {
          int row = r + 8 * laneHi;
          float val = (g[nt][r] + b2) * sInv[wave][row];
          atomicAdd(&out2[(size_t)sBase[wave][row] * 64 + col], val);
        }
      }
    }
    __syncthreads();
  }
}

extern "C" void kernel_launch(void* const* d_in, const int* in_sizes, int n_in,
                              void* d_out, int out_size, void* d_ws, size_t ws_size,
                              hipStream_t stream) {
  const float* ns  = (const float*)d_in[0];
  const float* nc  = (const float*)d_in[1];
  const float* nv  = (const float*)d_in[2];
  const float* pos = (const float*)d_in[3];
  const long long* eidx = (const long long*)d_in[4];
  const long long* tix  = (const long long*)d_in[5];
  const float* Ws1 = (const float*)d_in[6];
  const float* bs1 = (const float*)d_in[7];
  const float* Ws2 = (const float*)d_in[8];
  const float* bs2 = (const float*)d_in[9];
  const float* Wc1 = (const float*)d_in[10];
  const float* bc1 = (const float*)d_in[11];
  const float* Wc2 = (const float*)d_in[12];
  const float* bc2 = (const float*)d_in[13];
  const float* WV  = (const float*)d_in[14];
  const float* bV  = (const float*)d_in[15];

  int N = in_sizes[0] / 64;
  int E = in_sizes[4] / 2;
  int T = in_sizes[5] / 4;

  float* out1 = (float*)d_out;
  float* out2 = out1 + (size_t)N * 64;
  float* out3 = out2 + (size_t)N * 64;

  // Initialize outputs with residual inputs (atomics/RMW accumulate on top).
  hipMemcpyAsync(out1, ns, (size_t)N * 64  * sizeof(float), hipMemcpyDeviceToDevice, stream);
  hipMemcpyAsync(out2, nc, (size_t)N * 64  * sizeof(float), hipMemcpyDeviceToDevice, stream);
  hipMemcpyAsync(out3, nv, (size_t)N * 192 * sizeof(float), hipMemcpyDeviceToDevice, stream);

  int tilesE = (E + 15) / 16;
  int blocksE = (tilesE + WAVES * TPW - 1) / (WAVES * TPW);
  k_edge<<<blocksE, 256, 0, stream>>>(ns, nv, pos, eidx, Ws1, bs1, Ws2, bs2, out1, out3, tilesE);

  int tilesN = (N + 15) / 16;
  int blocksN = (tilesN + WAVES - 1) / WAVES;
  k_vs<<<blocksN, 256, 0, stream>>>(ns, nv, WV, bV, out1, tilesN);

  int tilesT = (T + 15) / 16;
  int blocksT = (tilesT + WAVES * TPW - 1) / (WAVES * TPW);
  k_tri<<<blocksT, 256, 0, stream>>>(nc, pos, tix, Wc1, bc1, Wc2, bc2, out2, tilesT);
}
